// MotionGuidedWindowAttention_71184787964113
// MI455X (gfx1250) — matrix operand order repaired and checked by reading, not compiled, not run on hardware
//
#include <hip/hip_runtime.h>
#include <hip/hip_bf16.h>

typedef __bf16 bf16_t;
typedef __attribute__((ext_vector_type(16))) __bf16        v16bf;
typedef __attribute__((ext_vector_type(8)))  float         v8f;
typedef __attribute__((ext_vector_type(4)))  unsigned int  u32x4;

#define DIM    256
#define NHEADS 8
#define HDIM   32
#define NTOK   64
#define ATT_SCALE 0.17677669529663687f   // 32^-0.5

// ---- LDS layout (bytes) ----
// X  [64x256 bf16]  0      .. 32767   (dead after QKV; reused for P)
// Q  [64x256 bf16]  32768  .. 65535   (dead after S;   reused for P)
// K  [64x256 bf16]  65536  .. 98303   (dead after S;   reused for attn_out)
// Vt [256x64 bf16]  98304  .. 131071  (V transposed: [d][token])
// bias [8x64 f32]   131072 .. 133119
#define X_OFF    0
#define Q_OFF    32768
#define K_OFF    65536
#define VT_OFF   98304
#define BIAS_OFF 131072
#define SMEM_BYTES 133120

struct __align__(8) bf16x4 { bf16_t a, b, c, d; };

__device__ __forceinline__ v8f zero8() {
  v8f z = {0.f, 0.f, 0.f, 0.f, 0.f, 0.f, 0.f, 0.f};
  return z;
}

__device__ __forceinline__ v16bf load_frag_2x8(const bf16_t* p0, const bf16_t* p1) {
  union { v16bf v; u32x4 u[2]; } u;
  u.u[0] = *(const u32x4*)p0;
  u.u[1] = *(const u32x4*)p1;
  return u.v;
}

// A fragment: 16x32 bf16, source row-major [M][K], row stride in elements.
// lane l<16: row l,    K chunks [0..7] and [16..23]
// lane l>=16: row l-16, K chunks [8..15] and [24..31]
__device__ __forceinline__ v16bf load_frag_a(const bf16_t* tile, int stride) {
  const int l = threadIdx.x & 31;
  const bf16_t* p = tile + (l & 15) * stride + ((l & 16) ? 8 : 0);
  return load_frag_2x8(p, p + 16);
}

// B fragment: 32x16 bf16, source is B^T row-major [N][K] (weight rows / K rows / Vt rows).
// lane l<16: N row l,    K chunks [0..7] and [16..23]
// lane l>=16: N row l-16, K chunks [8..15] and [24..31]
__device__ __forceinline__ v16bf load_frag_b(const bf16_t* tile, int stride) {
  const int l = threadIdx.x & 31;
  const bf16_t* p = tile + (l & 15) * stride + ((l & 16) ? 8 : 0);
  return load_frag_2x8(p, p + 16);
}

__device__ __forceinline__ v8f wmma_bf16(v16bf a, v16bf b, v8f c) {
  return __builtin_amdgcn_wmma_f32_16x16x32_bf16(false, a, false, b, (short)0, c,
                                                 false, false);
}

// ---------------- weight prep: f32 -> bf16 ----------------
__global__ void prep_weights(const float* __restrict__ qw, const float* __restrict__ pw,
                             bf16_t* __restrict__ wq, bf16_t* __restrict__ wp) {
  const int i = blockIdx.x * blockDim.x + threadIdx.x;
  if (i < 768 * 256 / 4) {
    const float4 t = ((const float4*)qw)[i];
    bf16x4 o = {(bf16_t)t.x, (bf16_t)t.y, (bf16_t)t.z, (bf16_t)t.w};
    ((bf16x4*)wq)[i] = o;
  }
  if (i < 256 * 256 / 4) {
    const float4 t = ((const float4*)pw)[i];
    bf16x4 o = {(bf16_t)t.x, (bf16_t)t.y, (bf16_t)t.z, (bf16_t)t.w};
    ((bf16x4*)wp)[i] = o;
  }
}

// ---------------- fused attention: 1 block = 1 window ----------------
__global__ __launch_bounds__(256)
void mgwa_kernel(const float* __restrict__ x,
                 const float* __restrict__ flow,
                 const bf16_t* __restrict__ wq,     // [768][256] bf16
                 const float* __restrict__ qkv_b,   // [768]
                 const bf16_t* __restrict__ wp,     // [256][256] bf16
                 const float* __restrict__ proj_b,  // [256]
                 const float* __restrict__ off_w1,  // [32][2]
                 const float* __restrict__ off_b1,  // [32]
                 const float* __restrict__ off_w2,  // [8][32]
                 const float* __restrict__ off_b2,  // [8]
                 float* __restrict__ out) {
  __shared__ __align__(16) unsigned char smem[SMEM_BYTES];

  const int b    = blockIdx.x;
  const int tid  = threadIdx.x;
  const int wave = tid >> 5;          // 0..7 (== head id in attention stages)
  const int l    = tid & 31;
  const int n16  = l & 15;            // N/col index inside a 16-wide tile
  const int g8   = (l & 16) ? 8 : 0;  // C/D row-group offset

  bf16_t* Xs = (bf16_t*)(smem + X_OFF);
  bf16_t* Qs = (bf16_t*)(smem + Q_OFF);
  bf16_t* Ks = (bf16_t*)(smem + K_OFF);
  bf16_t* Vt = (bf16_t*)(smem + VT_OFF);
  float*  Bs = (float*)(smem + BIAS_OFF);

  // ---- stage x tile as bf16 (vectorized: b128 load -> b64 LDS store) ----
  {
    const float4* xv = (const float4*)(x + (size_t)b * NTOK * DIM);
    bf16x4* xd = (bf16x4*)Xs;
    #pragma unroll
    for (int it = 0; it < (NTOK * DIM / 4) / 256; ++it) {
      const int i = it * 256 + tid;
      const float4 t = xv[i];
      bf16x4 o = {(bf16_t)t.x, (bf16_t)t.y, (bf16_t)t.z, (bf16_t)t.w};
      xd[i] = o;
    }
  }

  // ---- motion-bias MLP: wave == head, 2 rows per lane ----
  {
    const int h = wave;
    const float w2b = off_b2[h];
    for (int rr = l; rr < NTOK; rr += 32) {
      const float f0 = flow[(size_t)b * NTOK * 2 + rr * 2 + 0];
      const float f1 = flow[(size_t)b * NTOK * 2 + rr * 2 + 1];
      float acc = w2b;
      for (int j = 0; j < 32; ++j) {
        float hj = fmaxf(f0 * off_w1[2 * j] + f1 * off_w1[2 * j + 1] + off_b1[j], 0.f);
        acc += hj * off_w2[h * 32 + j];
      }
      Bs[h * NTOK + rr] = acc;
    }
  }
  __syncthreads();

  // ---- QKV GEMM: [64x256] x [256x768] ; each wave owns 6 of 48 N-tiles ----
  for (int nt = wave; nt < 48; nt += 8) {
    const int o0 = nt * 16;  // output column (0..767)
    v8f acc[4] = {zero8(), zero8(), zero8(), zero8()};
    for (int kt = 0; kt < 8; ++kt) {
      v16bf bfr = load_frag_b(wq + o0 * DIM + kt * 32, DIM);
      #pragma unroll
      for (int mt = 0; mt < 4; ++mt) {
        v16bf afr = load_frag_a(Xs + mt * 16 * DIM + kt * 32, DIM);
        acc[mt] = wmma_bf16(afr, bfr, acc[mt]);
      }
    }
    const float bias = qkv_b[o0 + n16];
    #pragma unroll
    for (int mt = 0; mt < 4; ++mt) {
      #pragma unroll
      for (int r = 0; r < 8; ++r) {
        const float v = acc[mt][r] + bias;
        const int m = mt * 16 + g8 + r;
        if (o0 < 256) {
          Qs[m * DIM + o0 + n16] = (bf16_t)v;
        } else if (o0 < 512) {
          Ks[m * DIM + (o0 - 256) + n16] = (bf16_t)v;
        } else {
          Vt[(o0 - 512 + n16) * NTOK + m] = (bf16_t)v;  // transposed store
        }
      }
    }
  }
  __syncthreads();

  // ---- S = Q K^T per head (wave == head); one WMMA per tile (K = 32 = HDIM) ----
  const int h = wave;
  v8f s[16];  // [mt*4 + nt]
  {
    v16bf bk[4];
    #pragma unroll
    for (int nt2 = 0; nt2 < 4; ++nt2) {
      bk[nt2] = load_frag_b(Ks + nt2 * 16 * DIM + h * HDIM, DIM);
    }
    #pragma unroll
    for (int mt = 0; mt < 4; ++mt) {
      v16bf aq = load_frag_a(Qs + mt * 16 * DIM + h * HDIM, DIM);
      #pragma unroll
      for (int nt2 = 0; nt2 < 4; ++nt2) {
        s[mt * 4 + nt2] = wmma_bf16(aq, bk[nt2], zero8());
      }
    }
  }

  // ---- scale + motion bias + softmax (row reductions across 16-lane groups) ----
  #pragma unroll
  for (int mt = 0; mt < 4; ++mt) {
    #pragma unroll
    for (int r = 0; r < 8; ++r) {
      const int row = mt * 16 + g8 + r;
      const float bb = Bs[h * NTOK + row];
      float mx = -3.0e38f;
      #pragma unroll
      for (int nt2 = 0; nt2 < 4; ++nt2) {
        const float v = s[mt * 4 + nt2][r] * ATT_SCALE + bb;
        s[mt * 4 + nt2][r] = v;
        mx = fmaxf(mx, v);
      }
      #pragma unroll
      for (int msk = 1; msk <= 8; msk <<= 1) mx = fmaxf(mx, __shfl_xor(mx, msk, 32));
      float sum = 0.f;
      #pragma unroll
      for (int nt2 = 0; nt2 < 4; ++nt2) {
        const float e = __expf(s[mt * 4 + nt2][r] - mx);
        s[mt * 4 + nt2][r] = e;
        sum += e;
      }
      #pragma unroll
      for (int msk = 1; msk <= 8; msk <<= 1) sum += __shfl_xor(sum, msk, 32);
      const float inv = 1.0f / sum;
      #pragma unroll
      for (int nt2 = 0; nt2 < 4; ++nt2) s[mt * 4 + nt2][r] *= inv;
    }
  }
  __syncthreads();  // all waves past Q/K reads -> reuse [0, 64K) for P

  // ---- write P (bf16) into per-head region overlaying X+Q ----
  bf16_t* Pb = (bf16_t*)(smem + h * (NTOK * NTOK * 2));  // 8 KB per head
  #pragma unroll
  for (int mt = 0; mt < 4; ++mt) {
    #pragma unroll
    for (int nt2 = 0; nt2 < 4; ++nt2) {
      #pragma unroll
      for (int r = 0; r < 8; ++r) {
        Pb[(mt * 16 + g8 + r) * NTOK + nt2 * 16 + n16] = (bf16_t)s[mt * 4 + nt2][r];
      }
    }
  }

  // ---- O = P V per head: M=64, N=32, K=64 ----
  v8f oacc[8] = {zero8(), zero8(), zero8(), zero8(), zero8(), zero8(), zero8(), zero8()};
  #pragma unroll
  for (int kt = 0; kt < 2; ++kt) {
    v16bf bv0 = load_frag_b(Vt + (h * HDIM + 0)  * NTOK + kt * 32, NTOK);
    v16bf bv1 = load_frag_b(Vt + (h * HDIM + 16) * NTOK + kt * 32, NTOK);
    #pragma unroll
    for (int mt = 0; mt < 4; ++mt) {
      v16bf ap = load_frag_a(Pb + mt * 16 * NTOK + kt * 32, NTOK);
      oacc[mt * 2 + 0] = wmma_bf16(ap, bv0, oacc[mt * 2 + 0]);
      oacc[mt * 2 + 1] = wmma_bf16(ap, bv1, oacc[mt * 2 + 1]);
    }
  }

  // ---- attention output (bf16) overlays dead K region ----
  bf16_t* AO = (bf16_t*)(smem + K_OFF);  // [64][256]
  #pragma unroll
  for (int mt = 0; mt < 4; ++mt) {
    #pragma unroll
    for (int d2 = 0; d2 < 2; ++d2) {
      #pragma unroll
      for (int r = 0; r < 8; ++r) {
        AO[(mt * 16 + g8 + r) * DIM + h * HDIM + d2 * 16 + n16] =
            (bf16_t)oacc[mt * 2 + d2][r];
      }
    }
  }
  __syncthreads();

  // ---- projection GEMM: [64x256] x [256x256] ; 2 N-tiles per wave ----
  float* ob = out + (size_t)b * NTOK * DIM;
  #pragma unroll
  for (int nti = 0; nti < 2; ++nti) {
    const int c0 = (wave * 2 + nti) * 16;
    v8f acc[4] = {zero8(), zero8(), zero8(), zero8()};
    #pragma unroll
    for (int kt = 0; kt < 8; ++kt) {
      v16bf bfr = load_frag_b(wp + c0 * DIM + kt * 32, DIM);
      #pragma unroll
      for (int mt = 0; mt < 4; ++mt) {
        v16bf afr = load_frag_a(AO + mt * 16 * DIM + kt * 32, DIM);
        acc[mt] = wmma_bf16(afr, bfr, acc[mt]);
      }
    }
    const float pb = proj_b[c0 + n16];
    #pragma unroll
    for (int mt = 0; mt < 4; ++mt) {
      #pragma unroll
      for (int r = 0; r < 8; ++r) {
        ob[(mt * 16 + g8 + r) * DIM + c0 + n16] = acc[mt][r] + pb;
      }
    }
  }
}

extern "C" void kernel_launch(void* const* d_in, const int* in_sizes, int n_in,
                              void* d_out, int out_size, void* d_ws, size_t ws_size,
                              hipStream_t stream) {
  (void)in_sizes; (void)n_in; (void)out_size; (void)ws_size;
  const float* x       = (const float*)d_in[0];
  const float* flow    = (const float*)d_in[1];
  const float* qkv_w   = (const float*)d_in[2];
  const float* qkv_b   = (const float*)d_in[3];
  const float* proj_w  = (const float*)d_in[4];
  const float* proj_b  = (const float*)d_in[5];
  const float* off_w1  = (const float*)d_in[6];
  const float* off_b1  = (const float*)d_in[7];
  const float* off_w2  = (const float*)d_in[8];
  const float* off_b2  = (const float*)d_in[9];

  bf16_t* wq = (bf16_t*)d_ws;             // 768*256 bf16
  bf16_t* wp = wq + 768 * 256;            // 256*256 bf16

  prep_weights<<<(768 * 256 / 4 + 255) / 256, 256, 0, stream>>>(qkv_w, proj_w, wq, wp);
  mgwa_kernel<<<8192, 256, 0, stream>>>(x, flow, wq, qkv_b, wp, proj_b,
                                        off_w1, off_b1, off_w2, off_b2,
                                        (float*)d_out);
}
